// MILSTM_70403103916354
// MI455X (gfx1250) — matrix operand order repaired and checked by reading, not compiled, not run on hardware
//
#include <hip/hip_runtime.h>
#include <hip/hip_bf16.h>

#define B_  32
#define T_  512
#define H_  1024
#define L_  2
#define FH_ 4096
#define KSPLIT_ 2
#define KHALF_  (H_ / KSPLIT_)   // 512
#define KSTEPS_ (KHALF_ / 32)    // 16

typedef __attribute__((ext_vector_type(16))) __bf16 bf16x16;
typedef __attribute__((ext_vector_type(8)))  __bf16 bf16x8;
typedef __attribute__((ext_vector_type(8)))  float  f32x8;

union FragAB { bf16x16 v; bf16x8 h[2]; };

struct Frags {
  FragAB ax, ah, bw0, bw1, bu0, bu1;
};

// ---------------------------------------------------------------- conversion
__global__ void milstm_cvt_f32_bf16(const float* __restrict__ src,
                                    __bf16* __restrict__ dst, int n) {
  int i = blockIdx.x * blockDim.x + threadIdx.x;
  if (i < n) dst[i] = (__bf16)src[i];
}

__global__ void milstm_zero_state(float* __restrict__ c,
                                  __bf16* __restrict__ h, int n) {
  int i = blockIdx.x * blockDim.x + threadIdx.x;
  if (i < n) { c[i] = 0.0f; h[i] = (__bf16)0.0f; }
}

// ---------------------------------------------------------------- GEMM stage
__device__ __forceinline__ void load_frags(
    Frags& f, int k0, int half,
    const __bf16* __restrict__ aRow, const __bf16* __restrict__ hRow,
    const __bf16* __restrict__ w0,  const __bf16* __restrict__ w1,
    const __bf16* __restrict__ u0,  const __bf16* __restrict__ u1) {
  // A fragment (16x32 bf16): lanes 0-15 K=[0..7],[16..23];
  // lanes 16-31 K=[8..15],[24..31] -> two contiguous 16B chunks.
  f.ax.h[0] = *(const bf16x8*)(aRow + k0 + half * 8);
  f.ax.h[1] = *(const bf16x8*)(aRow + k0 + half * 8 + 16);
  f.ah.h[0] = *(const bf16x8*)(hRow + k0 + half * 8);
  f.ah.h[1] = *(const bf16x8*)(hRow + k0 + half * 8 + 16);
  // B fragment (32x16 bf16): lanes 0-15 K=[0..7],[16..23]; lanes 16-31 K=[8..15],[24..31];
  // W row-major (n,k): B[k][n] = W[n][k] -> two contiguous 16B chunks per lane.
  f.bw0.h[0] = *(const bf16x8*)(w0 + k0 + half * 8);
  f.bw0.h[1] = *(const bf16x8*)(w0 + k0 + half * 8 + 16);
  f.bw1.h[0] = *(const bf16x8*)(w1 + k0 + half * 8);
  f.bw1.h[1] = *(const bf16x8*)(w1 + k0 + half * 8 + 16);
  f.bu0.h[0] = *(const bf16x8*)(u0 + k0 + half * 8);
  f.bu0.h[1] = *(const bf16x8*)(u0 + k0 + half * 8 + 16);
  f.bu1.h[0] = *(const bf16x8*)(u1 + k0 + half * 8);
  f.bu1.h[1] = *(const bf16x8*)(u1 + k0 + half * 8 + 16);
}

__device__ __forceinline__ void do_wmma(const Frags& f, f32x8& x0, f32x8& x1,
                                        f32x8& h0, f32x8& h1) {
  x0 = __builtin_amdgcn_wmma_f32_16x16x32_bf16(
      false, f.ax.v, false, f.bw0.v, (short)0, x0, false, false);
  x1 = __builtin_amdgcn_wmma_f32_16x16x32_bf16(
      false, f.ax.v, false, f.bw1.v, (short)0, x1, false, false);
  h0 = __builtin_amdgcn_wmma_f32_16x16x32_bf16(
      false, f.ah.v, false, f.bu0.v, (short)0, h0, false, false);
  h1 = __builtin_amdgcn_wmma_f32_16x16x32_bf16(
      false, f.ah.v, false, f.bu1.v, (short)0, h1, false, false);
}

// Partial-sum GEMM: gx_part = inp @ W^T, gh_part = h @ U^T over one K-half.
// Grid: 64 blocks = 32 N-blocks x 2 K-halves, 256 threads (8 waves). Each
// wave owns a 16(M) x 32(N) strip = two 16x16 tiles sharing the same A
// fragments -> 4 WMMAs per 12 b128 loads per k-step. Explicit double-buffered
// software pipeline: loads for step kb+1 issue before WMMAs of step kb.
__global__ __launch_bounds__(256) void milstm_gemm_partial(
    const __bf16* __restrict__ A, int lda,        // (B_ x H_) activations
    const __bf16* __restrict__ Hst,               // (B_ x H_) hidden, stride H_
    const __bf16* __restrict__ Wl,                // (FH_ x H_) row-major
    const __bf16* __restrict__ Ul,                // (FH_ x H_) row-major
    float*        __restrict__ gxp,               // (KSPLIT_ x B_ x FH_)
    float*        __restrict__ ghp)               // (KSPLIT_ x B_ x FH_)
{
  const int nb    = blockIdx.x & 31;          // N-block: 128 columns
  const int kh    = blockIdx.x >> 5;          // K-half: 512 of 1024
  const int lane  = threadIdx.x & 31;
  const int wave  = threadIdx.x >> 5;
  const int mtile = wave & 1;                 // M = 32 -> two 16-tiles
  const int wt    = wave >> 1;                // 0..3: 32-column strip in block
  const int half  = lane >> 4;                // lane half selects K sub-block
  const int l15   = lane & 15;
  const int am    = mtile * 16 + l15;         // A-matrix row (batch index)
  const int n0    = nb * 128 + wt * 32;
  const int bn0   = n0 + l15;                 // B column, first tile
  const int bn1   = bn0 + 16;                 // B column, second tile
  const int kbase = kh * KHALF_;

  f32x8 accX0 = {}, accX1 = {};
  f32x8 accH0 = {}, accH1 = {};

  const __bf16* aRow = A   + (size_t)am  * lda + kbase;
  const __bf16* hRow = Hst + (size_t)am  * H_  + kbase;
  const __bf16* w0   = Wl  + (size_t)bn0 * H_  + kbase;
  const __bf16* w1   = Wl  + (size_t)bn1 * H_  + kbase;
  const __bf16* u0   = Ul  + (size_t)bn0 * H_  + kbase;
  const __bf16* u1   = Ul  + (size_t)bn1 * H_  + kbase;

  Frags fA, fB;
  load_frags(fA, 0, half, aRow, hRow, w0, w1, u0, u1);

#pragma unroll
  for (int kb = 0; kb < KSTEPS_; ++kb) {
    if ((kb & 1) == 0) {
      if (kb + 1 < KSTEPS_)
        load_frags(fB, (kb + 1) * 32, half, aRow, hRow, w0, w1, u0, u1);
      do_wmma(fA, accX0, accX1, accH0, accH1);
    } else {
      if (kb + 1 < KSTEPS_)
        load_frags(fA, (kb + 1) * 32, half, aRow, hRow, w0, w1, u0, u1);
      do_wmma(fB, accX0, accX1, accH0, accH1);
    }
  }

  // C/D layout: VGPR r -> row (half*8 + r), col = lane&15.
  float* gxOut = gxp + (size_t)kh * B_ * FH_;
  float* ghOut = ghp + (size_t)kh * B_ * FH_;
#pragma unroll
  for (int r = 0; r < 8; ++r) {
    const int mm = mtile * 16 + half * 8 + r;
    gxOut[(size_t)mm * FH_ + bn0] = accX0[r];
    gxOut[(size_t)mm * FH_ + bn1] = accX1[r];
    ghOut[(size_t)mm * FH_ + bn0] = accH0[r];
    ghOut[(size_t)mm * FH_ + bn1] = accH1[r];
  }
}

// ------------------------------------------------------------- gating stage
__device__ __forceinline__ float sigm_(float x) {
  return 1.0f / (1.0f + __expf(-x));
}

// One block per batch row. Sums the two K-half partials, forms MI gates
// (gx+gh+gx*gh) into LDS once, then LN(4096), LSTM cell update, LN(1024),
// h = sigmoid(o)*tanh(c). Writes bf16 hidden state (+ optional f32 output).
__global__ __launch_bounds__(256) void milstm_gate_update(
    const float* __restrict__ gxp,         // (KSPLIT_ x B_ x FH_)
    const float* __restrict__ ghp,         // (KSPLIT_ x B_ x FH_)
    const float* __restrict__ bias,        // (FH_)
    float*       __restrict__ c,           // (B_ x H_) cell state (in/out)
    __bf16*      __restrict__ hbf,         // (B_ x H_) hidden state (out)
    const float* __restrict__ lngw, const float* __restrict__ lngb,
    const float* __restrict__ lncw, const float* __restrict__ lncb,
    float*       __restrict__ yout)        // d_out + t*H_ (stride T_*H_) or null
{
  __shared__ float r1[256];
  __shared__ float r2[256];
  __shared__ float grow[FH_];   // 16 KB: combined MI gates
  __shared__ float crow[H_];    //  4 KB: pre-LN cell values

  const int b   = blockIdx.x;
  const int tid = threadIdx.x;
  const float* gx0 = gxp + (size_t)b * FH_;
  const float* gx1 = gxp + (size_t)(B_ + b) * FH_;
  const float* gh0 = ghp + (size_t)b * FH_;
  const float* gh1 = ghp + (size_t)(B_ + b) * FH_;

  // ---- combine K-halves + MI, stats over the 4096 gate pre-activations
  float s = 0.0f, s2 = 0.0f;
  for (int i = tid; i < FH_; i += 256) {
    const float gx = gx0[i] + gx1[i] + bias[i];
    const float gh = gh0[i] + gh1[i];
    const float v  = gx + gh + gx * gh;
    grow[i] = v; s += v; s2 += v * v;
  }
  r1[tid] = s; r2[tid] = s2; __syncthreads();
  for (int st = 128; st > 0; st >>= 1) {
    if (tid < st) { r1[tid] += r1[tid + st]; r2[tid] += r2[tid + st]; }
    __syncthreads();
  }
  const float mean = r1[0] * (1.0f / FH_);
  const float var  = r2[0] * (1.0f / FH_) - mean * mean;
  const float rstd = rsqrtf(var + 1e-5f);
  __syncthreads();

  // ---- i/f/g/o, cell update (pre-LN cell into LDS)
  float oval[4];
#pragma unroll
  for (int k = 0; k < 4; ++k) {
    const int j = tid + k * 256;
    const float gi = (grow[j          ] - mean) * rstd * lngw[j          ] + lngb[j          ];
    const float gf = (grow[j +     H_ ] - mean) * rstd * lngw[j +     H_ ] + lngb[j +     H_ ];
    const float gg = (grow[j + 2 * H_ ] - mean) * rstd * lngw[j + 2 * H_ ] + lngb[j + 2 * H_ ];
    const float go = (grow[j + 3 * H_ ] - mean) * rstd * lngw[j + 3 * H_ ] + lngb[j + 3 * H_ ];
    const float cp = c[(size_t)b * H_ + j];
    crow[j] = sigm_(gf) * cp + sigm_(gi) * tanhf(gg);
    oval[k] = go;
  }
  __syncthreads();

  // ---- stats over the 1024 cell values
  s = 0.0f; s2 = 0.0f;
#pragma unroll
  for (int k = 0; k < 4; ++k) { float v = crow[tid + k * 256]; s += v; s2 += v * v; }
  r1[tid] = s; r2[tid] = s2; __syncthreads();
  for (int st = 128; st > 0; st >>= 1) {
    if (tid < st) { r1[tid] += r1[tid + st]; r2[tid] += r2[tid + st]; }
    __syncthreads();
  }
  const float mean2 = r1[0] * (1.0f / H_);
  const float var2  = r2[0] * (1.0f / H_) - mean2 * mean2;
  const float rstd2 = rsqrtf(var2 + 1e-5f);
  __syncthreads();

#pragma unroll
  for (int k = 0; k < 4; ++k) {
    const int j = tid + k * 256;
    const float cn = (crow[j] - mean2) * rstd2 * lncw[j] + lncb[j];
    c[(size_t)b * H_ + j] = cn;
    const float h = sigm_(oval[k]) * tanhf(cn);
    hbf[(size_t)b * H_ + j] = (__bf16)h;
    if (yout) yout[(size_t)b * (T_ * H_) + j] = h;
  }
}

// --------------------------------------------------------------------- host
extern "C" void kernel_launch(void* const* d_in, const int* in_sizes, int n_in,
                              void* d_out, int out_size, void* d_ws, size_t ws_size,
                              hipStream_t stream) {
  (void)in_sizes; (void)n_in; (void)out_size; (void)ws_size;

  const float* x    = (const float*)d_in[0];   // (B,T,H)
  const float* W    = (const float*)d_in[1];   // (L,4H,H)
  const float* bia  = (const float*)d_in[2];   // (L,4H)
  const float* U    = (const float*)d_in[3];   // (L,4H,H)
  const float* lngw = (const float*)d_in[4];   // (L,4H)
  const float* lngb = (const float*)d_in[5];   // (L,4H)
  const float* lncw = (const float*)d_in[6];   // (L,H)
  const float* lncb = (const float*)d_in[7];   // (L,H)
  float* y = (float*)d_out;                    // (B,T,H)

  char* ws = (char*)d_ws;
  size_t off = 0;
  auto alloc = [&](size_t bytes) -> void* {
    void* p = ws + off;
    off += (bytes + 255) & ~(size_t)255;
    return p;
  };
  __bf16* xb  = (__bf16*)alloc((size_t)B_ * T_ * H_ * 2);       // x in bf16
  __bf16* Wb  = (__bf16*)alloc((size_t)L_ * FH_ * H_ * 2);      // W bf16 (L2)
  __bf16* Ub  = (__bf16*)alloc((size_t)L_ * FH_ * H_ * 2);      // U bf16 (L2)
  __bf16* hb  = (__bf16*)alloc((size_t)L_ * B_ * H_ * 2);       // hidden state
  float*  cs  = (float*) alloc((size_t)L_ * B_ * H_ * 4);       // cell state
  float*  gxp = (float*) alloc((size_t)KSPLIT_ * B_ * FH_ * 4); // gx partials
  float*  ghp = (float*) alloc((size_t)KSPLIT_ * B_ * FH_ * 4); // gh partials

  int n;
  n = B_ * T_ * H_;
  milstm_cvt_f32_bf16<<<(n + 255) / 256, 256, 0, stream>>>(x, xb, n);
  n = L_ * FH_ * H_;
  milstm_cvt_f32_bf16<<<(n + 255) / 256, 256, 0, stream>>>(W, Wb, n);
  milstm_cvt_f32_bf16<<<(n + 255) / 256, 256, 0, stream>>>(U, Ub, n);
  n = L_ * B_ * H_;
  milstm_zero_state<<<(n + 255) / 256, 256, 0, stream>>>(cs, hb, n);

  for (int t = 0; t < T_; ++t) {
    for (int l = 0; l < L_; ++l) {
      const __bf16* A   = (l == 0) ? (xb + (size_t)t * H_)
                                   : (hb + (size_t)(l - 1) * B_ * H_);
      const int     lda = (l == 0) ? (T_ * H_) : H_;
      milstm_gemm_partial<<<32 * KSPLIT_, 256, 0, stream>>>(
          A, lda,
          hb + (size_t)l * B_ * H_,
          Wb + (size_t)l * FH_ * H_,
          Ub + (size_t)l * FH_ * H_,
          gxp, ghp);
      milstm_gate_update<<<B_, 256, 0, stream>>>(
          gxp, ghp,
          bia + (size_t)l * FH_,
          cs + (size_t)l * B_ * H_,
          hb + (size_t)l * B_ * H_,
          lngw + (size_t)l * FH_, lngb + (size_t)l * FH_,
          lncw + (size_t)l * H_,  lncb + (size_t)l * H_,
          (l == L_ - 1) ? (y + (size_t)t * H_) : nullptr);
    }
  }
}
